// VAE_88459146428838
// MI455X (gfx1250) — matrix operand order repaired and checked by reading, not compiled, not run on hardware
//
#include <hip/hip_runtime.h>
#include <hip/hip_bf16.h>
#include <math.h>

// ---------------------------------------------------------------------------
// VAE seq2seq (LSTM enc/dec, H=2048, T=256, V=128, L=256, C=8) for gfx1250.
// - bf16 WMMA (v_wmma_f32_16x16x32_bf16) for all big matmuls.
// - Recurrent matvec runs on WMMA too (B = broadcast h); it is L2-bandwidth
//   bound (32 MB Whh/step, resident in 192 MB L2), so the 16x column
//   redundancy is free.
// - h vector is staged into LDS once per block via TENSOR_LOAD_TO_LDS (TDM),
//   completed with s_wait_tensorcnt; an opaque asm memory-clobber publishes
//   the TDM-written LDS to the compiler so ds_load reads materialize.
// ---------------------------------------------------------------------------

typedef __attribute__((ext_vector_type(16))) __bf16       v16bf;
typedef __attribute__((ext_vector_type(8)))  __bf16       bf16x8;
typedef __attribute__((ext_vector_type(8)))  float        v8f;
typedef __attribute__((ext_vector_type(4)))  unsigned int v4u;
typedef __attribute__((ext_vector_type(8)))  int          v8i;
typedef __attribute__((ext_vector_type(4)))  int          v4i;

#define HD   2048
#define GD   8192      // 4*HD
#define TT   256
#define VV   128
#define LL   256
#define CC   8

__device__ __forceinline__ float sigmoidf_(float x) { return 1.0f / (1.0f + expf(-x)); }

#if __has_builtin(__builtin_amdgcn_tensor_load_to_lds)
#define HAVE_TDM 1
// 1-D TDM load: n_elems bf16 from global gsrc -> LDS offset lds_off.
// D# bitfields per CDNA5 ISA:
//   g0: [1:0]=count=1 | lds_addr[31:0] | global_addr[56:0] | type=2
//   g1: data_size=2B (code 1), tensor_dim0=n, tensor_dim1=1, tile_dim0=n,
//       tile_dim1=1, stride0=n
__device__ __forceinline__ void tdm_load_1d_to_lds(const void* gsrc, unsigned lds_off, unsigned n) {
    unsigned long long ga = (unsigned long long)gsrc;
    v4u g0;
    g0[0] = 1u;                                            // count=1, user mode
    g0[1] = lds_off;                                       // lds_addr (bytes)
    g0[2] = (unsigned)(ga & 0xffffffffu);                  // global_addr[31:0]
    g0[3] = (unsigned)((ga >> 32) & 0x01ffffffu) | (2u << 30); // addr[56:32] | type=2
    v8i g1;
    g1[0] = (int)(1u << 16);                               // data_size = 2 bytes
    g1[1] = (int)((n & 0xffffu) << 16);                    // tensor_dim0[15:0]
    g1[2] = (int)(((n >> 16) & 0xffffu) | (1u << 16));     // tensor_dim0[31:16] | tensor_dim1=1
    g1[3] = (int)((n & 0xffffu) << 16);                    // tensor_dim1 hi=0 | tile_dim0=n
    g1[4] = 1;                                             // tile_dim1=1, tile_dim2=0
    g1[5] = (int)n;                                        // tensor_dim0_stride[31:0]
    g1[6] = 0;
    g1[7] = 0;
    v4i z4 = {0, 0, 0, 0};
#if defined(__clang_major__) && (__clang_major__ >= 23)
    v8i z8 = {0, 0, 0, 0, 0, 0, 0, 0};
    __builtin_amdgcn_tensor_load_to_lds(g0, g1, z4, z4, z8, 0);
#else
    __builtin_amdgcn_tensor_load_to_lds(g0, g1, z4, z4, 0);
#endif
}
#else
#define HAVE_TDM 0
#endif

// ---------------- f32 -> bf16 bulk convert ----------------
__global__ void cvt_bf16(const float* __restrict__ in, __bf16* __restrict__ out, unsigned long n) {
    unsigned long i = (unsigned long)blockIdx.x * blockDim.x + threadIdx.x;
    if (i < n) out[i] = (__bf16)in[i];
}

// ---------------- embedding gather (+SOS shift, +relu) -> bf16 ----------------
__global__ void gather_embed(const int* __restrict__ words, const float* __restrict__ emb,
                             __bf16* __restrict__ out, int H, int shift_sos, int do_relu) {
    int t = blockIdx.x;
    int w = shift_sos ? ((t == 0) ? 0 : words[t - 1]) : words[t];
    const float* row = emb + (size_t)w * H;
    for (int j = threadIdx.x; j < H; j += blockDim.x) {
        float v = row[j];
        if (do_relu) v = fmaxf(v, 0.0f);
        out[(size_t)t * H + j] = (__bf16)v;
    }
}

// ---------------- WMMA GEMM: Out[M,N] = A[M,K] * W[N,K]^T + bias0 + bias1 ----------------
// block = 128 threads (4 waves), each wave computes one 16x16 tile; block covers 16x64.
__global__ void gemm_bf16_wmma(const __bf16* __restrict__ A, const __bf16* __restrict__ W,
                               const float* __restrict__ bias0, const float* __restrict__ bias1,
                               float* __restrict__ Out, int M, int N, int K) {
    const int lane = threadIdx.x & 31;
    const int wv   = threadIdx.x >> 5;
    const int m0 = blockIdx.y * 16;
    const int n0 = (blockIdx.x * 4 + wv) * 16;
    if (m0 >= M || n0 >= N) return;          // wave-uniform exit (EXEC stays full)
    const int mr = lane & 15;
    const int hf = lane >> 4;

    const __bf16* arow = A + (size_t)(m0 + mr) * K;
    const __bf16* wrow = W + (size_t)(n0 + mr) * K;
    v8f acc = {0.f,0.f,0.f,0.f,0.f,0.f,0.f,0.f};

    for (int k0 = 0; k0 < K; k0 += 32) {
        // A fragment: row m=l%16; elems 0..7 -> K=hf*8+e ; 8..15 -> K=16+hf*8+e
        bf16x8 a_lo = *(const bf16x8*)(arow + k0 + hf * 8);
        bf16x8 a_hi = *(const bf16x8*)(arow + k0 + 16 + hf * 8);
        // B fragment: col n=l%16; elems 0..7 -> K=hf*8+e ; 8..15 -> K=16+hf*8+e  (B col n == W row n0+n)
        bf16x8 b_lo = *(const bf16x8*)(wrow + k0 + hf * 8);
        bf16x8 b_hi = *(const bf16x8*)(wrow + k0 + 16 + hf * 8);
        v16bf a, b;
        #pragma unroll
        for (int e = 0; e < 8; ++e) { a[e] = a_lo[e]; a[e+8] = a_hi[e]; b[e] = b_lo[e]; b[e+8] = b_hi[e]; }
        acc = __builtin_amdgcn_wmma_f32_16x16x32_bf16(false, a, false, b, (short)0, acc, false, false);
    }
    const int n = n0 + mr;
    float bias = 0.0f;
    if (bias0) bias += bias0[n];
    if (bias1) bias += bias1[n];
    #pragma unroll
    for (int e = 0; e < 8; ++e) {
        int m = m0 + hf * 8 + e;             // D: lane col n=l%16, rows (l/16)*8+e
        Out[(size_t)m * N + n] = acc[e] + bias;
    }
}

// ---------------- recurrent gates: gates[R] = Whh[R,K] @ h + Gx[t] ----------------
// One wave per 16 rows. h staged to LDS once per block via TDM; B fragments are
// broadcast ds reads. Whh streams from L2 with a one-block-ahead prefetch hint.
__global__ void lstm_gates_wmma(const __bf16* __restrict__ Whh, const __bf16* __restrict__ hb,
                                const float* __restrict__ gx, float* __restrict__ gates,
                                int K, int R) {
    __shared__ __bf16 shH[HD];              // single LDS object -> offset 0

#if HAVE_TDM
    if ((threadIdx.x >> 5) == 0) {          // one TDM issue per block (wave op, EXEC ignored)
        tdm_load_1d_to_lds(hb, 0u, (unsigned)K);
        __builtin_amdgcn_s_wait_tensorcnt(0);
    }
    // TDM wrote LDS behind the compiler's back: escape the LDS pointer and
    // clobber memory so the ds_load reads below must be materialized.
    asm volatile("" : : "r"(&shH[0]) : "memory");
#else
    for (int j = threadIdx.x; j < K; j += blockDim.x) shH[j] = hb[j];
#endif
    __syncthreads();

    const int lane = threadIdx.x & 31;
    const int wv = blockIdx.x * (blockDim.x >> 5) + (threadIdx.x >> 5);
    const int r0 = wv * 16;
    if (r0 >= R) return;
    const int mr = lane & 15;
    const int hf = lane >> 4;
    const __bf16* arow = Whh + (size_t)(r0 + mr) * K;
    v8f acc = {0.f,0.f,0.f,0.f,0.f,0.f,0.f,0.f};

    for (int k0 = 0; k0 < K; k0 += 32) {
        __builtin_prefetch(arow + k0 + 1024, 0, 0);              // next-block hint (speculative)
        bf16x8 a_lo = *(const bf16x8*)(arow + k0 + hf * 8);
        bf16x8 a_hi = *(const bf16x8*)(arow + k0 + 16 + hf * 8);
        bf16x8 h_lo = *(const bf16x8*)(shH + k0 + hf * 8);       // LDS broadcast
        bf16x8 h_hi = *(const bf16x8*)(shH + k0 + 16 + hf * 8);
        v16bf a, b;
        #pragma unroll
        for (int e = 0; e < 8; ++e) { a[e] = a_lo[e]; a[e+8] = a_hi[e]; b[e] = h_lo[e]; b[e+8] = h_hi[e]; }
        acc = __builtin_amdgcn_wmma_f32_16x16x32_bf16(false, a, false, b, (short)0, acc, false, false);
    }
    if (mr == 0) {                           // column 0 of D holds the matvec result
        #pragma unroll
        for (int e = 0; e < 8; ++e) {
            int r = r0 + hf * 8 + e;
            gates[r] = acc[e] + gx[r];
        }
    }
}

// ---------------- LSTM cell elementwise update ----------------
__global__ void lstm_cell(const float* __restrict__ gates, float* __restrict__ h, float* __restrict__ c,
                          __bf16* __restrict__ hb, __bf16* __restrict__ hseq_b, int H) {
    int j = blockIdx.x * blockDim.x + threadIdx.x;
    if (j >= H) return;
    float ig = sigmoidf_(gates[j]);
    float fg = sigmoidf_(gates[H + j]);
    float gg = tanhf(gates[2 * H + j]);
    float og = sigmoidf_(gates[3 * H + j]);
    float cn = fg * c[j] + ig * gg;
    float hn = og * tanhf(cn);
    c[j] = cn;
    h[j] = hn;
    hb[j] = (__bf16)hn;
    if (hseq_b) hseq_b[j] = (__bf16)hn;
}

__global__ void init_state(const float* __restrict__ h0, const float* __restrict__ c0,
                           float* __restrict__ h, float* __restrict__ c, __bf16* __restrict__ hb, int H) {
    int j = blockIdx.x * blockDim.x + threadIdx.x;
    if (j >= H) return;
    h[j] = h0[j]; c[j] = c0[j]; hb[j] = (__bf16)h0[j];
}

__global__ void state_to_bf16(const float* __restrict__ h, __bf16* __restrict__ hb, int H) {
    int j = blockIdx.x * blockDim.x + threadIdx.x;
    if (j < H) hb[j] = (__bf16)h[j];
}

// ---------------- small f32 matvec: out[r] = W[r,:K] . x + b[r] (1 wave/row) ----------------
__global__ void matvec_f32(const float* __restrict__ Wm, const float* __restrict__ x,
                           const float* __restrict__ b, float* __restrict__ out, int R, int K) {
    int row = (blockIdx.x * blockDim.x + threadIdx.x) >> 5;
    int lane = threadIdx.x & 31;
    if (row >= R) return;
    const float* wr = Wm + (size_t)row * K;
    float s = 0.0f;
    for (int k = lane; k < K; k += 32) s += wr[k] * x[k];
    #pragma unroll
    for (int off = 16; off; off >>= 1) s += __shfl_xor(s, off, 32);
    if (lane == 0) out[row] = s + b[row];
}

// ---------------- reparameterize + concat cond + KL ----------------
__global__ void reparam_kl(const float* __restrict__ mh, const float* __restrict__ lvh, const float* __restrict__ eh,
                           const float* __restrict__ mc, const float* __restrict__ lvc, const float* __restrict__ ec,
                           const float* __restrict__ cond_emb, const int* __restrict__ tense,
                           float* __restrict__ zh, float* __restrict__ zc, float* __restrict__ kl_out) {
    __shared__ float red[256];
    int l = threadIdx.x;
    float kl = 0.0f;
    if (l < LL) {
        float a = mh[l], bb = lvh[l];
        zh[l] = a + eh[l] * expf(0.5f * bb);
        kl += -0.5f * (1.0f + bb - a * a - expf(bb));
        float a2 = mc[l], b2 = lvc[l];
        zc[l] = a2 + ec[l] * expf(0.5f * b2);
        kl += -0.5f * (1.0f + b2 - a2 * a2 - expf(b2));
    }
    if (l < CC) {
        float cv = cond_emb[tense[0] * CC + l];
        zh[LL + l] = cv;
        zc[LL + l] = cv;
    }
    red[l] = kl;
    __syncthreads();
    for (int off = 128; off; off >>= 1) { if (l < off) red[l] += red[l + off]; __syncthreads(); }
    if (l == 0) kl_out[0] = red[0];
}

// ---------------- cross entropy over logits [T,V] ----------------
__global__ void ce_kernel(const float* __restrict__ logits, const int* __restrict__ tgt,
                          float* __restrict__ out_ce, int T, int V) {
    __shared__ float red[256];
    int t = threadIdx.x;
    float nll = 0.0f;
    if (t < T) {
        const float* row = logits + (size_t)t * V;
        float m = -1e30f;
        for (int j = 0; j < V; ++j) m = fmaxf(m, row[j]);
        float s = 0.0f;
        for (int j = 0; j < V; ++j) s += expf(row[j] - m);
        nll = (m + logf(s)) - row[tgt[t]];
    }
    red[t] = nll;
    __syncthreads();
    for (int off = 128; off; off >>= 1) { if (t < off) red[t] += red[t + off]; __syncthreads(); }
    if (t == 0) out_ce[0] = red[0] / (float)T;
}

// ---------------------------------------------------------------------------
extern "C" void kernel_launch(void* const* d_in, const int* in_sizes, int n_in,
                              void* d_out, int out_size, void* d_ws, size_t ws_size,
                              hipStream_t stream) {
    (void)in_sizes; (void)n_in; (void)out_size; (void)ws_size;
    const int*   input_words  = (const int*)d_in[0];
    const int*   target_words = (const int*)d_in[1];
    const int*   tense        = (const int*)d_in[2];
    const float* h0      = (const float*)d_in[3];
    const float* c0      = (const float*)d_in[4];
    const float* eps_h   = (const float*)d_in[5];
    const float* eps_c   = (const float*)d_in[6];
    const float* enc_emb = (const float*)d_in[7];
    const float* enc_Wih = (const float*)d_in[8];
    const float* enc_Whh = (const float*)d_in[9];
    const float* enc_bih = (const float*)d_in[10];
    const float* enc_bhh = (const float*)d_in[11];
    const float* dec_emb = (const float*)d_in[12];
    const float* dec_Wih = (const float*)d_in[13];
    const float* dec_Whh = (const float*)d_in[14];
    const float* dec_bih = (const float*)d_in[15];
    const float* dec_bhh = (const float*)d_in[16];
    const float* dec_Wout= (const float*)d_in[17];
    const float* dec_bout= (const float*)d_in[18];
    const float* W_h2m  = (const float*)d_in[19];  const float* b_h2m  = (const float*)d_in[20];
    const float* W_h2lv = (const float*)d_in[21];  const float* b_h2lv = (const float*)d_in[22];
    const float* W_c2m  = (const float*)d_in[23];  const float* b_c2m  = (const float*)d_in[24];
    const float* W_c2lv = (const float*)d_in[25];  const float* b_c2lv = (const float*)d_in[26];
    const float* W_l2h  = (const float*)d_in[27];  const float* b_l2h  = (const float*)d_in[28];
    const float* W_l2c  = (const float*)d_in[29];  const float* b_l2c  = (const float*)d_in[30];
    const float* cond_emb = (const float*)d_in[31];
    float* out = (float*)d_out;                      // out[0]=CE, out[1]=KL

    // ---- workspace carve (~148 MB) ----
    char* p = (char*)d_ws;
    auto carve = [&](size_t bytes) -> void* {
        void* q = (void*)p;
        p += (bytes + 255) & ~(size_t)255;
        return q;
    };
    __bf16* encWih_b = (__bf16*)carve((size_t)GD * HD * 2);
    __bf16* encWhh_b = (__bf16*)carve((size_t)GD * HD * 2);
    __bf16* decWih_b = (__bf16*)carve((size_t)GD * HD * 2);
    __bf16* decWhh_b = (__bf16*)carve((size_t)GD * HD * 2);
    __bf16* Wout_b   = (__bf16*)carve((size_t)VV * HD * 2);
    __bf16* enc_xs_b = (__bf16*)carve((size_t)TT * HD * 2);
    __bf16* dec_xs_b = (__bf16*)carve((size_t)TT * HD * 2);
    float*  Gx_enc   = (float*)carve((size_t)TT * GD * 4);
    float*  Gx_dec   = (float*)carve((size_t)TT * GD * 4);
    __bf16* dec_hs_b = (__bf16*)carve((size_t)TT * HD * 2);
    float*  logits   = (float*)carve((size_t)TT * VV * 4);
    float*  gates    = (float*)carve((size_t)GD * 4);
    float*  hbuf     = (float*)carve((size_t)HD * 4);
    float*  cbuf     = (float*)carve((size_t)HD * 4);
    __bf16* hb       = (__bf16*)carve((size_t)HD * 2);
    float*  mean_h   = (float*)carve(LL * 4);
    float*  logvar_h = (float*)carve(LL * 4);
    float*  mean_c   = (float*)carve(LL * 4);
    float*  logvar_c = (float*)carve(LL * 4);
    float*  zcat_h   = (float*)carve((LL + CC) * 4);
    float*  zcat_c   = (float*)carve((LL + CC) * 4);

    const unsigned long nW = (unsigned long)GD * HD;   // 16.7M elems
    const int CB = 256;

    // ---- one-time bf16 weight conversion (Whh pair = 64 MB bf16, resident in 192 MB L2) ----
    cvt_bf16<<<(unsigned)((nW + CB - 1) / CB), CB, 0, stream>>>(enc_Wih, encWih_b, nW);
    cvt_bf16<<<(unsigned)((nW + CB - 1) / CB), CB, 0, stream>>>(enc_Whh, encWhh_b, nW);
    cvt_bf16<<<(unsigned)((nW + CB - 1) / CB), CB, 0, stream>>>(dec_Wih, decWih_b, nW);
    cvt_bf16<<<(unsigned)((nW + CB - 1) / CB), CB, 0, stream>>>(dec_Whh, decWhh_b, nW);
    cvt_bf16<<<(unsigned)(((unsigned long)VV * HD + CB - 1) / CB), CB, 0, stream>>>(dec_Wout, Wout_b, (unsigned long)VV * HD);

    // ---- embedding gathers ----
    gather_embed<<<TT, 256, 0, stream>>>(input_words, enc_emb, enc_xs_b, HD, 0, 0);
    gather_embed<<<TT, 256, 0, stream>>>(target_words, dec_emb, dec_xs_b, HD, 1, 1);  // [SOS, tgt[:-1]], relu

    // ---- batched x-gate GEMMs: Gx[t,r] = x_t . Wih[r,:] + bih[r] + bhh[r] ----
    gemm_bf16_wmma<<<dim3(GD / 64, TT / 16), 128, 0, stream>>>(enc_xs_b, encWih_b, enc_bih, enc_bhh, Gx_enc, TT, GD, HD);
    gemm_bf16_wmma<<<dim3(GD / 64, TT / 16), 128, 0, stream>>>(dec_xs_b, decWih_b, dec_bih, dec_bhh, Gx_dec, TT, GD, HD);

    // ---- encoder scan ----
    init_state<<<HD / 256, 256, 0, stream>>>(h0, c0, hbuf, cbuf, hb, HD);
    for (int t = 0; t < TT; ++t) {
        lstm_gates_wmma<<<GD / 16 / 8, 256, 0, stream>>>(encWhh_b, hb, Gx_enc + (size_t)t * GD, gates, HD, GD);
        lstm_cell<<<HD / 256, 256, 0, stream>>>(gates, hbuf, cbuf, hb, (__bf16*)nullptr, HD);
    }

    // ---- latent heads (read h_enc/c_enc from hbuf/cbuf) ----
    matvec_f32<<<LL / 8, 256, 0, stream>>>(W_h2m,  hbuf, b_h2m,  mean_h,   LL, HD);
    matvec_f32<<<LL / 8, 256, 0, stream>>>(W_h2lv, hbuf, b_h2lv, logvar_h, LL, HD);
    matvec_f32<<<LL / 8, 256, 0, stream>>>(W_c2m,  cbuf, b_c2m,  mean_c,   LL, HD);
    matvec_f32<<<LL / 8, 256, 0, stream>>>(W_c2lv, cbuf, b_c2lv, logvar_c, LL, HD);
    reparam_kl<<<1, 256, 0, stream>>>(mean_h, logvar_h, eps_h, mean_c, logvar_c, eps_c,
                                      cond_emb, tense, zcat_h, zcat_c, out + 1);

    // ---- decoder initial state: dec_h0 = W_l2h @ [z_h; cond] + b ----
    matvec_f32<<<HD / 8, 256, 0, stream>>>(W_l2h, zcat_h, b_l2h, hbuf, HD, LL + CC);
    matvec_f32<<<HD / 8, 256, 0, stream>>>(W_l2c, zcat_c, b_l2c, cbuf, HD, LL + CC);
    state_to_bf16<<<HD / 256, 256, 0, stream>>>(hbuf, hb, HD);

    // ---- decoder scan (stores h_t in bf16 for the logits GEMM) ----
    for (int t = 0; t < TT; ++t) {
        lstm_gates_wmma<<<GD / 16 / 8, 256, 0, stream>>>(decWhh_b, hb, Gx_dec + (size_t)t * GD, gates, HD, GD);
        lstm_cell<<<HD / 256, 256, 0, stream>>>(gates, hbuf, cbuf, hb, dec_hs_b + (size_t)t * HD, HD);
    }

    // ---- logits = dec_hs @ Wout^T + bout, then CE ----
    gemm_bf16_wmma<<<dim3(VV / 64, TT / 16), 128, 0, stream>>>(dec_hs_b, Wout_b, dec_bout, (const float*)nullptr,
                                                               logits, TT, VV, HD);
    ce_kernel<<<1, 256, 0, stream>>>(logits, target_words, out, TT, VV);
}
